// Model_4002909520791
// MI455X (gfx1250) — compile-verified, matrix operand order checked by reading, not run on hardware
//
#include <hip/hip_runtime.h>
#include <hip/hip_bf16.h>

#define TT 1024
#define EMBD 300
#define KE 320          // EMB padded to multiple of 32 for WMMA K
#define HID 512
#define G4 2048         // 4*HID gates
#define H2 1024         // 2*HID
#define TAGSET 22
#define TAGP 32         // padded tagset for WMMA N tiles
#define START_TAG 20
#define END_TAG 21
#define NEGINF (-10000.0f)
#define MAXK 1024       // max K for the LDS B-tile (32KB)

typedef _Float16 v16h __attribute__((ext_vector_type(16)));
typedef _Float16 v8h  __attribute__((ext_vector_type(8)));
typedef float    v8f  __attribute__((ext_vector_type(8)));
typedef unsigned int v4u __attribute__((ext_vector_type(4)));
typedef unsigned int v8u __attribute__((ext_vector_type(8)));

union V16 { v16h v; v8h h[2]; };

// ---------------- fp32 -> fp16 convert with zero padding ----------------
__global__ void cvt_f16_pad(const float* __restrict__ src, _Float16* __restrict__ dst,
                            int srows, int scols, int drows, int dcols) {
  int idx = blockIdx.x * blockDim.x + threadIdx.x;
  int n = drows * dcols;
  if (idx >= n) return;
  int r = idx / dcols, c = idx % dcols;
  float v = (r < srows && c < scols) ? src[(size_t)r * scols + c] : 0.0f;
  dst[idx] = (_Float16)v;
}

// ---------------- embedding gather -> fp16 padded (T x KE) ----------------
__global__ void embed_gather(const int* __restrict__ ids, const float* __restrict__ emb,
                             _Float16* __restrict__ x) {
  int idx = blockIdx.x * blockDim.x + threadIdx.x;
  if (idx >= TT * KE) return;
  int t = idx / KE, k = idx % KE;
  float v = (k < EMBD) ? emb[(size_t)ids[t] * EMBD + k] : 0.0f;
  x[idx] = (_Float16)v;
}

// ---------------- WMMA GEMM with TDM-staged B tile ----------------
// C[M,N] = A[M,K] * B[N,K]^T + bias.  Grid (M/64, N/16), 128 threads = 4 waves.
// Wave w computes M-tile (blockIdx.x*4 + w) x N-tile blockIdx.y.
// The 16-row B tile (16 x K fp16) is DMA'd into LDS once per block by the
// Tensor Data Mover (2D descriptor, ISA 8.3-8.6), then read via ds_load_b128.
__global__ void __launch_bounds__(128) wmma_gemm_nt(
    const _Float16* __restrict__ A, const _Float16* __restrict__ B,
    const float* __restrict__ bias, int n_bias,
    float* __restrict__ C, int K, int ldc) {
  __shared__ _Float16 Bsh[16 * MAXK];

  const int tid  = threadIdx.x;
  const int lane = tid & 31;
  const int wv   = tid >> 5;            // wave 0..3
  const int mt   = blockIdx.x * 4 + wv;
  const int nt   = blockIdx.y;
  const int r16  = lane & 15;
  const int hi   = lane >> 4;           // 0 or 1
  const int sel  = hi ? 8 : 0;          // K sub-chunk select

  // ---- TDM: stage B tile rows [nt*16, nt*16+16) x K into LDS (wave 0 only)
  if (tid < 32) {
    unsigned long long ga = (unsigned long long)(const void*)(B + (size_t)nt * 16 * K);
    unsigned int ldsOff = (unsigned int)(size_t)(&Bsh[0]);
    v4u g0;
    g0[0] = 1u;                                   // count=1, user descriptor
    g0[1] = ldsOff;                               // lds_addr
    g0[2] = (unsigned int)ga;                     // global_addr[31:0]
    g0[3] = (unsigned int)(ga >> 32) | 0x80000000u; // global_addr[56:32] | type=2
    v8u g1;
    g1[0] = 0x10000u;                             // data_size=1 (2 bytes)
    g1[1] = ((unsigned int)K & 0xFFFFu) << 16;    // tensor_dim0[15:0]
    g1[2] = ((unsigned int)K >> 16) | (16u << 16);// tensor_dim0[31:16] | tensor_dim1=16
    g1[3] = ((unsigned int)K) << 16;              // tile_dim0 = K (halves)
    g1[4] = 16u;                                  // tile_dim1 = 16, tile_dim2 = 0
    g1[5] = (unsigned int)K;                      // tensor_dim0_stride[31:0]
    g1[6] = 0u;                                   // stride hi / dim1_stride lo
    g1[7] = 0u;
    v4u g2 = (v4u)(0u);
    v4u g3 = (v4u)(0u);
    asm volatile("tensor_load_to_lds %0, %1, %2, %3"
                 :: "s"(g0), "s"(g1), "s"(g2), "s"(g3) : "memory");
    __builtin_amdgcn_s_wait_tensorcnt(0);
  }
  __syncthreads();

  const _Float16* rowA = A + (size_t)(mt * 16 + r16) * K;
  const _Float16* rowB = &Bsh[(size_t)r16 * K];
  v8f acc = {};
  for (int k0 = 0; k0 < K; k0 += 32) {
    V16 a, b;
    a.h[0] = *(const v8h*)(rowA + k0 + sel);
    a.h[1] = *(const v8h*)(rowA + k0 + sel + 16);
    b.h[0] = *(const v8h*)(rowB + k0 + sel);        // ds_load_b128
    b.h[1] = *(const v8h*)(rowB + k0 + sel + 16);
    if (k0 + 64 <= K)                               // prefetch next A K-tile
      __builtin_prefetch(rowA + k0 + 32, 0, 0);
    acc = __builtin_amdgcn_wmma_f32_16x16x32_f16(
        /*neg_a=*/false, a.v, /*neg_b=*/false, b.v,
        /*c_mod=*/(short)0, acc, /*reuse_a=*/false, /*reuse_b=*/false);
  }
  int n = nt * 16 + r16;
  float bv = (bias != nullptr && n < n_bias) ? bias[n] : 0.0f;
  // C/D layout: VGPR r, lane l -> M = r + 8*(l>=16), N = l&15
  float* outp = C + (size_t)(mt * 16 + hi * 8) * ldc + n;
#pragma unroll
  for (int r = 0; r < 8; ++r)
    outp[(size_t)r * ldc] = acc[r] + bv;
}

// ---------------- persistent bidirectional LSTM scan ----------------
__device__ __forceinline__ float sigmoidf_(float x) { return 1.0f / (1.0f + expf(-x)); }

// blockIdx.x = 0 (forward) / 1 (backward); 1024 threads.
// pre: (T, 2048) precomputed x@W_ih^T + b; whh fp16 (2048, 512), L2-resident.
// Thread j computes gate elements j and j+1024 of the 2048-vector per step.
__global__ void __launch_bounds__(1024) lstm_bidir(
    const float* __restrict__ preF, const float* __restrict__ preB,
    const _Float16* __restrict__ whhF, const _Float16* __restrict__ whhB,
    const float* __restrict__ h0, const float* __restrict__ c0,
    float* __restrict__ outH) {
  const int dir = blockIdx.x;
  const float* pre = dir ? preB : preF;
  const _Float16* whh = dir ? whhB : whhF;
  __shared__ float h_sh[HID];
  __shared__ float g_sh[G4];
  const int j = threadIdx.x;
  if (j < HID) h_sh[j] = h0[dir * HID + j];
  float c = (j < HID) ? c0[dir * HID + j] : 0.0f;
  __syncthreads();
  const _Float16* w0 = whh + (size_t)j * HID;          // 1024B-aligned rows
  const _Float16* w1 = whh + (size_t)(j + H2) * HID;
  for (int s = 0; s < TT; ++s) {
    const int t = dir ? (TT - 1 - s) : s;
    const float* p = pre + (size_t)t * G4;
    float a0 = p[j], a1 = p[j + H2];
#pragma unroll 4
    for (int k = 0; k < HID; k += 8) {
      v8h wv0 = *(const v8h*)(w0 + k);                 // global_load_b128
      v8h wv1 = *(const v8h*)(w1 + k);
#pragma unroll
      for (int e = 0; e < 8; ++e) {
        float hk = h_sh[k + e];
        a0 = fmaf((float)wv0[e], hk, a0);
        a1 = fmaf((float)wv1[e], hk, a1);
      }
    }
    g_sh[j] = a0;
    g_sh[j + H2] = a1;
    __syncthreads();
    if (j < HID) {
      float ii = sigmoidf_(g_sh[j]);               // i gate
      float ff = sigmoidf_(g_sh[HID + j]);         // f gate
      float gg = tanhf(g_sh[H2 + j]);              // g gate
      float oo = sigmoidf_(g_sh[H2 + HID + j]);    // o gate
      c = ff * c + ii * gg;
      float h = oo * tanhf(c);
      h_sh[j] = h;
      outH[(size_t)t * H2 + dir * HID + j] = h;
    }
    __syncthreads();
  }
}

// ---------------- CRF: forward logsumexp + gold score + Viterbi ----------------
__global__ void __launch_bounds__(32) crf_kernel(
    const float* __restrict__ feats, const float* __restrict__ trans,
    const int* __restrict__ tags, int* __restrict__ bp, float* __restrict__ out) {
  __shared__ float tr[TAGSET * TAGSET];
  __shared__ float alpha[TAGSET];
  __shared__ float val[TAGSET];
  const int j = threadIdx.x;
  for (int i = j; i < TAGSET * TAGSET; i += 32) tr[i] = trans[i];
  if (j < TAGSET) {
    float a0 = (j == START_TAG) ? 0.0f : NEGINF;
    alpha[j] = a0;
    val[j] = a0;
  }
  __syncthreads();
  for (int t = 0; t < TT; ++t) {
    float na = 0.0f, nv = 0.0f;
    int arg = 0;
    if (j < TAGSET) {
      float f = feats[t * TAGP + j];
      float m = -3.4e38f;
      for (int i = 0; i < TAGSET; ++i) m = fmaxf(m, alpha[i] + tr[i * TAGSET + j]);
      float sum = 0.0f;
      for (int i = 0; i < TAGSET; ++i) sum += expf(alpha[i] + tr[i * TAGSET + j] - m);
      na = f + m + logf(sum);
      float bm = -3.4e38f;
      for (int i = 0; i < TAGSET; ++i) {
        float s = val[i] + tr[i * TAGSET + j];
        if (s > bm) { bm = s; arg = i; }
      }
      nv = bm + f;
      bp[t * TAGP + j] = arg;
    }
    __syncthreads();
    if (j < TAGSET) { alpha[j] = na; val[j] = nv; }
    __syncthreads();
  }
  if (j == 0) {
    // forward partition score
    float m = -3.4e38f;
    for (int i = 0; i < TAGSET; ++i) m = fmaxf(m, alpha[i] + tr[i * TAGSET + END_TAG]);
    float s = 0.0f;
    for (int i = 0; i < TAGSET; ++i) s += expf(alpha[i] + tr[i * TAGSET + END_TAG] - m);
    float fwd = m + logf(s);
    // gold path score
    float gold = 0.0f;
    int prev = START_TAG;
    for (int t = 0; t < TT; ++t) {
      int tg = tags[t];
      gold += tr[prev * TAGSET + tg] + feats[t * TAGP + tg];
      prev = tg;
    }
    gold += tr[prev * TAGSET + END_TAG];
    out[TT] = fwd - gold;
    // Viterbi backtrace
    float bm = -3.4e38f;
    int best = 0;
    for (int i = 0; i < TAGSET; ++i) {
      float s2 = val[i] + tr[i * TAGSET + END_TAG];
      if (s2 > bm) { bm = s2; best = i; }
    }
    int tag = best;
    out[TT - 1] = (float)tag;
    for (int t = TT - 1; t >= 1; --t) {
      tag = bp[t * TAGP + tag];
      out[t - 1] = (float)tag;
    }
  }
}

// ---------------- host launch ----------------
extern "C" void kernel_launch(void* const* d_in, const int* in_sizes, int n_in,
                              void* d_out, int out_size, void* d_ws, size_t ws_size,
                              hipStream_t stream) {
  (void)in_sizes; (void)n_in; (void)out_size; (void)ws_size;
  const int*   sent    = (const int*)d_in[0];
  const int*   tags    = (const int*)d_in[1];
  const float* emb     = (const float*)d_in[2];
  const float* e_wih_f = (const float*)d_in[3];
  const float* e_whh_f = (const float*)d_in[4];
  const float* e_b_f   = (const float*)d_in[5];
  const float* e_wih_b = (const float*)d_in[6];
  const float* e_whh_b = (const float*)d_in[7];
  const float* e_b_b   = (const float*)d_in[8];
  const float* dc_wih_f= (const float*)d_in[9];
  const float* dc_whh_f= (const float*)d_in[10];
  const float* dc_b_f  = (const float*)d_in[11];
  const float* dc_wih_b= (const float*)d_in[12];
  const float* dc_whh_b= (const float*)d_in[13];
  const float* dc_b_b  = (const float*)d_in[14];
  const float* lin_w   = (const float*)d_in[15];
  const float* lin_b   = (const float*)d_in[16];
  const float* trans   = (const float*)d_in[17];
  const float* h0      = (const float*)d_in[18];
  const float* c0      = (const float*)d_in[19];
  float* out = (float*)d_out;

  char* ws = (char*)d_ws;
  size_t off = 0;
  auto alloc = [&](size_t bytes) -> void* {
    void* p = ws + off;
    off = (off + bytes + 255) & ~(size_t)255;
    return p;
  };
  _Float16* xe16    = (_Float16*)alloc((size_t)TT * KE * 2);
  _Float16* ewihf16 = (_Float16*)alloc((size_t)G4 * KE * 2);
  _Float16* ewihb16 = (_Float16*)alloc((size_t)G4 * KE * 2);
  _Float16* ewhhf16 = (_Float16*)alloc((size_t)G4 * HID * 2);
  _Float16* ewhhb16 = (_Float16*)alloc((size_t)G4 * HID * 2);
  _Float16* dwihf16 = (_Float16*)alloc((size_t)G4 * H2 * 2);
  _Float16* dwihb16 = (_Float16*)alloc((size_t)G4 * H2 * 2);
  _Float16* dwhhf16 = (_Float16*)alloc((size_t)G4 * HID * 2);
  _Float16* dwhhb16 = (_Float16*)alloc((size_t)G4 * HID * 2);
  _Float16* lin16   = (_Float16*)alloc((size_t)TAGP * H2 * 2);
  float*    preF    = (float*)alloc((size_t)TT * G4 * 4);
  float*    preB    = (float*)alloc((size_t)TT * G4 * 4);
  float*    hEnc    = (float*)alloc((size_t)TT * H2 * 4);
  _Float16* hEnc16  = (_Float16*)alloc((size_t)TT * H2 * 2);
  float*    hDec    = (float*)alloc((size_t)TT * H2 * 4);
  _Float16* hDec16  = (_Float16*)alloc((size_t)TT * H2 * 2);
  float*    feats   = (float*)alloc((size_t)TT * TAGP * 4);
  int*      bp      = (int*)alloc((size_t)TT * TAGP * 4);

  dim3 b256(256);
  auto cvt = [&](const float* s, _Float16* d, int sr, int sc, int dr, int dc) {
    int n = dr * dc;
    cvt_f16_pad<<<dim3((n + 255) / 256), b256, 0, stream>>>(s, d, sr, sc, dr, dc);
  };

  // Stage inputs in fp16 (padded K for WMMA)
  embed_gather<<<dim3((TT * KE + 255) / 256), b256, 0, stream>>>(sent, emb, xe16);
  cvt(e_wih_f,  ewihf16, G4, EMBD, G4, KE);
  cvt(e_wih_b,  ewihb16, G4, EMBD, G4, KE);
  cvt(e_whh_f,  ewhhf16, G4, HID,  G4, HID);
  cvt(e_whh_b,  ewhhb16, G4, HID,  G4, HID);
  cvt(dc_wih_f, dwihf16, G4, H2,   G4, H2);
  cvt(dc_wih_b, dwihb16, G4, H2,   G4, H2);
  cvt(dc_whh_f, dwhhf16, G4, HID,  G4, HID);
  cvt(dc_whh_b, dwhhb16, G4, HID,  G4, HID);
  cvt(lin_w,    lin16,   TAGSET, H2, TAGP, H2);

  dim3 b128(128);
  // Encoder: batched input projections (WMMA + TDM-staged B), then scan
  wmma_gemm_nt<<<dim3(TT / 64, G4 / 16), b128, 0, stream>>>(xe16, ewihf16, e_b_f, G4, preF, KE, G4);
  wmma_gemm_nt<<<dim3(TT / 64, G4 / 16), b128, 0, stream>>>(xe16, ewihb16, e_b_b, G4, preB, KE, G4);
  lstm_bidir<<<dim3(2), dim3(1024), 0, stream>>>(preF, preB, ewhhf16, ewhhb16, h0, c0, hEnc);
  cvt(hEnc, hEnc16, TT, H2, TT, H2);

  // Decoder
  wmma_gemm_nt<<<dim3(TT / 64, G4 / 16), b128, 0, stream>>>(hEnc16, dwihf16, dc_b_f, G4, preF, H2, G4);
  wmma_gemm_nt<<<dim3(TT / 64, G4 / 16), b128, 0, stream>>>(hEnc16, dwihb16, dc_b_b, G4, preB, H2, G4);
  lstm_bidir<<<dim3(2), dim3(1024), 0, stream>>>(preF, preB, dwhhf16, dwhhb16, h0, c0, hDec);
  cvt(hDec, hDec16, TT, H2, TT, H2);

  // Emission features (WMMA) + CRF
  wmma_gemm_nt<<<dim3(TT / 64, TAGP / 16), b128, 0, stream>>>(hDec16, lin16, lin_b, TAGSET, feats, H2, TAGP);
  crf_kernel<<<dim3(1), dim3(32), 0, stream>>>(feats, trans, tags, bp, out);
}
